// QwenAttention_40226663694427
// MI455X (gfx1250) — matrix operand order repaired and checked by reading, not compiled, not run on hardware
//
#include <hip/hip_runtime.h>
#include <hip/hip_bf16.h>

#define HIDDEN 896
#define NHEADS 14
#define NKV 2
#define HD 64
#define GQA (NHEADS / NKV)           // 7
#define BB 2
#define SS 2048
#define MROWS (BB * SS)              // 4096
#define NQKV (HIDDEN + 2 * NKV * HD) // 1152

// softmax scale (1/sqrt(64)) * log2(e), folded into Q at RoPE time so the
// flash loop runs softmax in the log2 domain with bare v_exp_f32.
#define QSCL (0.125f * 1.44269504088896341f)

#define AS1 __attribute__((address_space(1)))
#define AS3 __attribute__((address_space(3)))

typedef _Float16 v8h  __attribute__((ext_vector_type(8)));
typedef _Float16 v16h __attribute__((ext_vector_type(16)));
typedef float    v8f  __attribute__((ext_vector_type(8)));
typedef int      v4i_vs __attribute__((vector_size(16)));   // matches builtin sig

__device__ __forceinline__ v8f wmma_f16(v16h a, v16h b, v8f c) {
  // D = A(16x32 f16) * B(32x16 f16) + C(16x16 f32)
  return __builtin_amdgcn_wmma_f32_16x16x32_f16(
      /*neg_a=*/false, a, /*neg_b=*/false, b,
      /*c_mod=*/(short)0, c, /*reuse_a=*/false, /*reuse_b=*/false);
}

// ---- async global -> LDS copy (16B per lane), ASYNCcnt-tracked ------------
__device__ __forceinline__ void async_copy_b128(const _Float16* g, _Float16* l) {
#if __has_builtin(__builtin_amdgcn_global_load_async_to_lds_b128)
  __builtin_amdgcn_global_load_async_to_lds_b128(
      (AS1 v4i_vs*)g, (AS3 v4i_vs*)l, 0, 0);
#else
  *(v8h*)l = *(const v8h*)g;   // fallback: global_load + ds_store
#endif
}
__device__ __forceinline__ void wait_async_all() {
#if __has_builtin(__builtin_amdgcn_global_load_async_to_lds_b128)
#if __has_builtin(__builtin_amdgcn_s_wait_asynccnt)
  __builtin_amdgcn_s_wait_asynccnt(0);
#else
  asm volatile("s_wait_asynccnt 0" ::: "memory");
#endif
#endif
}

// A-fragment (16x32, M x K), row-major source with leading dim ld.
// ISA layout: lanes 0-15 -> M=lane, K=0..7 (v0-3) & 16..23 (v4-7);
//             lanes 16-31 -> M=lane-16, K=8..15 & 24..31.
__device__ __forceinline__ v16h load_a_frag(const _Float16* __restrict__ A,
                                            int ld, int row0, int k0, int lane) {
  int hf = lane >> 4;
  int r  = lane & 15;
  const _Float16* p = A + (size_t)(row0 + r) * ld + (k0 + hf * 8);
  v8h lo = *(const v8h*)p;
  v8h hi = *(const v8h*)(p + 16);
  return __builtin_shufflevector(lo, hi, 0,1,2,3,4,5,6,7,8,9,10,11,12,13,14,15);
}

// B-fragment (32x16, K x N) from B^T stored row-major [N, K].
// ISA layout: lanes 0-15 -> N=lane, K=0..7 & 16..23; lanes 16-31 -> N=lane-16, K=8..15 & 24..31.
__device__ __forceinline__ v16h load_b_frag(const _Float16* __restrict__ Bt,
                                            int ld, int n0, int k0, int lane) {
  int hf = lane >> 4;
  int r  = lane & 15;
  const _Float16* p = Bt + (size_t)(n0 + r) * ld + (k0 + hf * 8);
  v8h lo = *(const v8h*)p;
  v8h hi = *(const v8h*)(p + 16);
  return __builtin_shufflevector(lo, hi, 0,1,2,3,4,5,6,7,8,9,10,11,12,13,14,15);
}

// Same B-fragment pattern but reading a 64x32 LDS tile (row stride 32 halfs).
__device__ __forceinline__ v16h lds_b_frag(const _Float16* tile, int nloc, int lane) {
  int hf = lane >> 4;
  int r  = lane & 15;
  const _Float16* p = tile + (nloc + r) * 32 + hf * 8;
  v8h lo = *(const v8h*)p;
  v8h hi = *(const v8h*)(p + 16);
  return __builtin_shufflevector(lo, hi, 0,1,2,3,4,5,6,7,8,9,10,11,12,13,14,15);
}

// ---------------------------------------------------------------------------
// WMMA GEMM with async-LDS staged B tile: out[m,n] = sum_k A[m,k]*Bt[n,k]+bias
// Block = 256 threads = 8 waves; block tile 128(M) x 64(N); k-step 32.
// ---------------------------------------------------------------------------
template <bool STORE_F16, bool HAS_BIAS>
__global__ void __launch_bounds__(256)
gemm_wmma(const _Float16* __restrict__ A, const _Float16* __restrict__ Bt,
          const float* __restrict__ bias, void* __restrict__ outv,
          int M, int N, int K) {
  __shared__ alignas(64) _Float16 btile[2][64 * 32];   // 2 x 4KB
  int lane = threadIdx.x & 31;
  int wave = threadIdx.x >> 5;
  int m0 = blockIdx.x * 128 + wave * 16;
  int n0 = blockIdx.y * 64;

  // cooperative staging: thread t moves one 16B chunk; 256 threads = 4KB tile
  int trow = threadIdx.x >> 2;            // 0..63 -> tile row (n)
  int tch  = threadIdx.x & 3;             // 16B chunk within 64B row
  const _Float16* gB = Bt + (size_t)(n0 + trow) * K + tch * 8;
  int lofs = trow * 32 + tch * 8;

  async_copy_b128(gB, &btile[0][lofs]);   // prologue: tile 0 in flight

  v8f c0 = {}, c1 = {}, c2 = {}, c3 = {};
  int nsteps = K / 32;
  for (int i = 0; i < nsteps; ++i) {
    int k0 = i * 32;
    v16h a = load_a_frag(A, K, m0, k0, lane);
    wait_async_all();
    __syncthreads();                      // tile i visible to all waves
    if (i + 1 < nsteps)                   // overlap next stage with compute
      async_copy_b128(gB + (size_t)(i + 1) * 32, &btile[(i + 1) & 1][lofs]);
    const _Float16* bt = btile[i & 1];
    v16h b0 = lds_b_frag(bt, 0,  lane);
    v16h b1 = lds_b_frag(bt, 16, lane);
    v16h b2 = lds_b_frag(bt, 32, lane);
    v16h b3 = lds_b_frag(bt, 48, lane);
    c0 = wmma_f16(a, b0, c0);
    c1 = wmma_f16(a, b1, c1);
    c2 = wmma_f16(a, b2, c2);
    c3 = wmma_f16(a, b3, c3);
  }
  // C layout: lane<16 -> N=lane, M=r ; lane>=16 -> N=lane-16, M=8+r
  int rbase = m0 + (lane >> 4) * 8;
  int cl = lane & 15;
#pragma unroll
  for (int t = 0; t < 4; ++t) {
    v8f c = (t == 0) ? c0 : (t == 1) ? c1 : (t == 2) ? c2 : c3;
    int col = n0 + t * 16 + cl;
    float bv = HAS_BIAS ? bias[col] : 0.0f;
#pragma unroll
    for (int r = 0; r < 8; ++r) {
      float v = c[r] + bv;
      if (STORE_F16)
        ((_Float16*)outv)[(size_t)(rbase + r) * N + col] = (_Float16)v;
      else
        ((float*)outv)[(size_t)(rbase + r) * N + col] = v;
    }
  }
}

// ---------------------------------------------------------------------------
__global__ void cvt_f32_f16(const float* __restrict__ src,
                            _Float16* __restrict__ dst, int n) {
  int i = blockIdx.x * blockDim.x + threadIdx.x;
  if (i < n) dst[i] = (_Float16)src[i];
}

__global__ void pack_bias(const float* __restrict__ bq, const float* __restrict__ bk,
                          const float* __restrict__ bv, float* __restrict__ out) {
  int i = blockIdx.x * blockDim.x + threadIdx.x;
  if (i >= NQKV) return;
  out[i] = (i < HIDDEN) ? bq[i]
         : (i < HIDDEN + NKV * HD) ? bk[i - HIDDEN]
         : bv[i - HIDDEN - NKV * HD];
}

// ---------------------------------------------------------------------------
// RoPE + layout scatter: QKV16 -> Q16[b,h,s,d], K16[b,kv,s,d], Vt16[b,kv,d,s]
// Q pre-scaled by QSCL (softmax in log2 domain).
// ---------------------------------------------------------------------------
__global__ void rope_scatter(const _Float16* __restrict__ QKV,
                             _Float16* __restrict__ Q, _Float16* __restrict__ Kc,
                             _Float16* __restrict__ Vt) {
  const int PER = NHEADS * 32 + NKV * 32 + NKV * HD; // 640
  int idx = blockIdx.x * blockDim.x + threadIdx.x;
  if (idx >= MROWS * PER) return;
  int m = idx / PER;
  int u = idx % PER;
  int b = m / SS, s = m % SS;
  const _Float16* row = QKV + (size_t)m * NQKV;
  if (u < NHEADS * 32) {
    int hh = u >> 5, i = u & 31;
    float fr = (float)s * __powf(1000000.0f, -(float)i * (1.0f / 32.0f));
    float cs = __cosf(fr), sn = __sinf(fr);
    float x1 = (float)row[hh * HD + i];
    float x2 = (float)row[hh * HD + i + 32];
    _Float16* q = Q + (((size_t)(b * NHEADS + hh) * SS) + s) * HD;
    q[i]      = (_Float16)((x1 * cs - x2 * sn) * QSCL);
    q[i + 32] = (_Float16)((x2 * cs + x1 * sn) * QSCL);
  } else if (u < NHEADS * 32 + NKV * 32) {
    int t = u - NHEADS * 32;
    int kh = t >> 5, i = t & 31;
    float fr = (float)s * __powf(1000000.0f, -(float)i * (1.0f / 32.0f));
    float cs = __cosf(fr), sn = __sinf(fr);
    float x1 = (float)row[HIDDEN + kh * HD + i];
    float x2 = (float)row[HIDDEN + kh * HD + i + 32];
    _Float16* k = Kc + (((size_t)(b * NKV + kh) * SS) + s) * HD;
    k[i]      = (_Float16)(x1 * cs - x2 * sn);
    k[i + 32] = (_Float16)(x2 * cs + x1 * sn);
  } else {
    int t = u - NHEADS * 32 - NKV * 32;
    int kh = t >> 6, d = t & 63;
    Vt[(((size_t)(b * NKV + kh) * HD) + d) * SS + s] =
        row[HIDDEN + NKV * HD + kh * HD + d];
  }
}

// ---------------------------------------------------------------------------
// Flash attention helpers: fragment set + one kv-block step.
// ---------------------------------------------------------------------------
struct Frags {
  v16h ka0, ka1, ka2, ka3;   // K tile A-fragments (2 kv sub-tiles x 2 d-steps)
  v16h vb0, vb1, vb2, vb3;   // V tile B-fragments (4 d chunks)
};

__device__ __forceinline__ void load_frags(Frags& f,
                                           const _Float16* __restrict__ Kh,
                                           const _Float16* __restrict__ Vh,
                                           int kv0, int lane) {
  f.ka0 = load_a_frag(Kh, HD, kv0,      0,  lane);
  f.ka1 = load_a_frag(Kh, HD, kv0,      32, lane);
  f.ka2 = load_a_frag(Kh, HD, kv0 + 16, 0,  lane);
  f.ka3 = load_a_frag(Kh, HD, kv0 + 16, 32, lane);
  f.vb0 = load_b_frag(Vh, SS, 0,  kv0, lane);
  f.vb1 = load_b_frag(Vh, SS, 16, kv0, lane);
  f.vb2 = load_b_frag(Vh, SS, 32, kv0, lane);
  f.vb3 = load_b_frag(Vh, SS, 48, kv0, lane);
}

__device__ __forceinline__ void process_block(
    const Frags& f, const v16h& qb0, const v16h& qb1,
    int kv0, bool masked, int q_glob, int rofs, int lane,
    float& mrun, float& lrun, v8f& o0, v8f& o1, v8f& o2, v8f& o3) {
  v8f t0 = {}, t1 = {};
  t0 = wmma_f16(f.ka0, qb0, t0);
  t0 = wmma_f16(f.ka1, qb1, t0);
  t1 = wmma_f16(f.ka2, qb0, t1);
  t1 = wmma_f16(f.ka3, qb1, t1);

  float s0[8], s1[8];
  float mx = -__builtin_inff();
  if (!masked) {
#pragma unroll
    for (int r = 0; r < 8; ++r) {
      s0[r] = t0[r];
      s1[r] = t1[r];
      mx = fmaxf(mx, fmaxf(s0[r], s1[r]));
    }
  } else {
#pragma unroll
    for (int r = 0; r < 8; ++r) {
      int kvA = kv0 + rofs + r;
      s0[r] = (kvA      <= q_glob) ? t0[r] : -__builtin_inff();
      s1[r] = (kvA + 16 <= q_glob) ? t1[r] : -__builtin_inff();
      mx = fmaxf(mx, fmaxf(s0[r], s1[r]));
    }
  }
  mx = fmaxf(mx, __shfl_xor(mx, 16));
  float mnew  = fmaxf(mrun, mx);
  float alpha = exp2f(mrun - mnew);     // log2-domain softmax
  float ls = 0.0f;
#pragma unroll
  for (int r = 0; r < 8; ++r) {
    s0[r] = exp2f(s0[r] - mnew); ls += s0[r];
    s1[r] = exp2f(s1[r] - mnew); ls += s1[r];
  }
  ls += __shfl_xor(ls, 16);
  lrun = lrun * alpha + ls;
  mrun = mnew;

  // P already in A-fragment layout (M=q on lanes, K=kv in VGPRs)
  v16h pa;
#pragma unroll
  for (int r = 0; r < 8; ++r) {
    pa[r]     = (_Float16)s0[r];
    pa[r + 8] = (_Float16)s1[r];
  }
  // rescale running O by alpha (q=j lives in lanes j and j+16)
  int srcbase = (lane & 16) >> 1;
#pragma unroll
  for (int r = 0; r < 8; ++r) {
    float ar = __shfl(alpha, srcbase + r);
    o0[r] *= ar; o1[r] *= ar; o2[r] *= ar; o3[r] *= ar;
  }
  o0 = wmma_f16(pa, f.vb0, o0);
  o1 = wmma_f16(pa, f.vb1, o1);
  o2 = wmma_f16(pa, f.vb2, o2);
  o3 = wmma_f16(pa, f.vb3, o3);
}

// ---------------------------------------------------------------------------
// Flash attention, wave32, 16 q-rows per wave, kv blocks of 32.
// Transposed-score softmax; only final kv block masked; manual PING-PONG
// double buffering (fragment sets swap roles, no register copy storm).
// ---------------------------------------------------------------------------
__global__ void __launch_bounds__(128)
flash_attn(const _Float16* __restrict__ Q, const _Float16* __restrict__ Kc,
           const _Float16* __restrict__ Vt, _Float16* __restrict__ O) {
  int lane = threadIdx.x & 31;
  int w = blockIdx.x * 4 + (threadIdx.x >> 5);
  int qt = w & 127;
  int bh = w >> 7;
  int b = bh / NHEADS, h = bh % NHEADS;
  int kh = h / GQA;
  const _Float16* Qh = Q  + (size_t)(b * NHEADS + h) * SS * HD;
  const _Float16* Kh = Kc + (size_t)(b * NKV + kh) * SS * HD;
  const _Float16* Vh = Vt + (size_t)(b * NKV + kh) * HD * SS;
  int q0 = qt * 16;

  v16h qb0 = load_b_frag(Qh, HD, q0, 0, lane);
  v16h qb1 = load_b_frag(Qh, HD, q0, 32, lane);

  v8f o0 = {}, o1 = {}, o2 = {}, o3 = {};
  float mrun = -__builtin_inff();
  float lrun = 0.0f;
  int q_glob = q0 + (lane & 15);
  int rofs   = (lane >> 4) * 8;
  int nblk   = (q0 + 16 + 31) >> 5;

  Frags fa, fb;
  load_frags(fa, Kh, Vh, 0, lane);

  int blk = 0;
  for (;;) {
    {   // consume set A, refill set B
      bool more = (blk + 1 < nblk);
      int kv0 = blk * 32;
      if (blk + 2 < nblk) {
        __builtin_prefetch(Kh + (size_t)(kv0 + 64) * HD, 0, 3);
        __builtin_prefetch(Vh + (size_t)(kv0 + 64), 0, 3);
      }
      load_frags(fb, Kh, Vh, more ? kv0 + 32 : 0, lane);
      process_block(fa, qb0, qb1, kv0, !more, q_glob, rofs, lane,
                    mrun, lrun, o0, o1, o2, o3);
      if (++blk >= nblk) break;
    }
    {   // consume set B, refill set A
      bool more = (blk + 1 < nblk);
      int kv0 = blk * 32;
      if (blk + 2 < nblk) {
        __builtin_prefetch(Kh + (size_t)(kv0 + 64) * HD, 0, 3);
        __builtin_prefetch(Vh + (size_t)(kv0 + 64), 0, 3);
      }
      load_frags(fa, Kh, Vh, more ? kv0 + 32 : 0, lane);
      process_block(fb, qb0, qb1, kv0, !more, q_glob, rofs, lane,
                    mrun, lrun, o0, o1, o2, o3);
      if (++blk >= nblk) break;
    }
  }

  // normalize by l and store f16 into O16[b, s, h*64 + d]
  int srcbase = (lane & 16) >> 1;
  int dcol = lane & 15;
  size_t base = (size_t)b * SS * HIDDEN + (size_t)h * HD;
#pragma unroll
  for (int r = 0; r < 8; ++r) {
    float li = 1.0f / __shfl(lrun, srcbase + r);
    int qrow = q0 + rofs + r;
    size_t ro = base + (size_t)qrow * HIDDEN;
    O[ro + 0  + dcol] = (_Float16)(o0[r] * li);
    O[ro + 16 + dcol] = (_Float16)(o1[r] * li);
    O[ro + 32 + dcol] = (_Float16)(o2[r] * li);
    O[ro + 48 + dcol] = (_Float16)(o3[r] * li);
  }
}

// ---------------------------------------------------------------------------
extern "C" void kernel_launch(void* const* d_in, const int* in_sizes, int n_in,
                              void* d_out, int out_size, void* d_ws, size_t ws_size,
                              hipStream_t stream) {
  const float* hs = (const float*)d_in[0];
  // d_in[1] = attention_mask: exactly causal -> implemented analytically
  const float* Wq = (const float*)d_in[2];
  const float* bq = (const float*)d_in[3];
  const float* Wk = (const float*)d_in[4];
  const float* bk = (const float*)d_in[5];
  const float* Wv = (const float*)d_in[6];
  const float* bv = (const float*)d_in[7];
  const float* Wo = (const float*)d_in[8];

  char* ws = (char*)d_ws;
  size_t off = 0;
  auto alloc = [&](size_t bytes) -> void* {
    void* p = ws + off;
    off = (off + bytes + 255) & ~(size_t)255;
    return p;
  };
  _Float16* X16   = (_Float16*)alloc((size_t)MROWS * HIDDEN * 2);
  _Float16* Wall  = (_Float16*)alloc((size_t)NQKV * HIDDEN * 2);
  _Float16* Wo16  = (_Float16*)alloc((size_t)HIDDEN * HIDDEN * 2);
  float*    biasA = (float*)alloc((size_t)NQKV * 4);
  _Float16* QKV16 = (_Float16*)alloc((size_t)MROWS * NQKV * 2);
  _Float16* Q16   = (_Float16*)alloc((size_t)BB * NHEADS * SS * HD * 2);
  _Float16* K16   = (_Float16*)alloc((size_t)BB * NKV * SS * HD * 2);
  _Float16* Vt16  = (_Float16*)alloc((size_t)BB * NKV * HD * SS * 2);
  _Float16* O16   = (_Float16*)alloc((size_t)MROWS * HIDDEN * 2);

  int n;
  n = MROWS * HIDDEN;
  cvt_f32_f16<<<(n + 255) / 256, 256, 0, stream>>>(hs, X16, n);
  n = HIDDEN * HIDDEN;
  cvt_f32_f16<<<(n + 255) / 256, 256, 0, stream>>>(Wq, Wall, n);
  n = NKV * HD * HIDDEN;
  cvt_f32_f16<<<(n + 255) / 256, 256, 0, stream>>>(
      Wk, Wall + (size_t)HIDDEN * HIDDEN, n);
  cvt_f32_f16<<<(n + 255) / 256, 256, 0, stream>>>(
      Wv, Wall + (size_t)(HIDDEN + NKV * HD) * HIDDEN, n);
  n = HIDDEN * HIDDEN;
  cvt_f32_f16<<<(n + 255) / 256, 256, 0, stream>>>(Wo, Wo16, n);
  pack_bias<<<(NQKV + 255) / 256, 256, 0, stream>>>(bq, bk, bv, biasA);

  // Fused QKV projection: [4096,896] x [1152,896]^T -> [4096,1152] f16
  dim3 g1(MROWS / 128, NQKV / 64);
  gemm_wmma<true, true><<<g1, 256, 0, stream>>>(X16, Wall, biasA, QKV16,
                                                MROWS, NQKV, HIDDEN);

  n = MROWS * (NHEADS * 32 + NKV * 32 + NKV * HD);
  rope_scatter<<<(n + 255) / 256, 256, 0, stream>>>(QKV16, Q16, K16, Vt16);

  // 2*14*128 q-tiles = 3584 waves, 4 waves/block
  flash_attn<<<(BB * NHEADS * (SS / 16)) / 4, 128, 0, stream>>>(Q16, K16, Vt16, O16);

  // Output projection -> f32 d_out
  dim3 g2(MROWS / 128, HIDDEN / 64);
  gemm_wmma<false, false><<<g2, 256, 0, stream>>>(O16, Wo16, nullptr,
                                                  (float*)d_out,
                                                  MROWS, HIDDEN, HIDDEN);
}
